// QKVAttention_12962211299457
// MI455X (gfx1250) — matrix operand order repaired and checked by reading, not compiled, not run on hardware
//
#include <hip/hip_runtime.h>
#include <hip/hip_bf16.h>
#include <math.h>

typedef __attribute__((ext_vector_type(16))) _Float16 v16h;
typedef __attribute__((ext_vector_type(8)))  _Float16 v8h;
typedef __attribute__((ext_vector_type(8)))  float    v8f;
typedef __attribute__((ext_vector_type(4)))  float    v4f;
typedef __attribute__((ext_vector_type(4)))  int      v4i;

#define NHEADS 16
#define CH     64
#define LEN    2048
#define BM     128          // M rows per workgroup (8 waves x 16)
#define BN     64           // S columns per tile
#define NTILES (LEN / BN)
#define LDSS   80           // padded LDS row stride in halves (160 B)

// scale = 1/sqrt(sqrt(64)); fold sqrt(log2(e)) into both q and k so that
// q*k products are already in log2 domain -> softmax uses native v_exp_f32.
#define QSCALE (0.35355339059327379f * 1.2011224087864498f)

#if defined(__AMDGCN__) && __has_builtin(__builtin_amdgcn_global_load_async_to_lds_b128)
#define HAVE_ASYNC 1
#else
#define HAVE_ASYNC 0
#endif

#if HAVE_ASYNC
# if __has_builtin(__builtin_amdgcn_s_wait_asynccnt)
#  define WAIT_ASYNC(n) __builtin_amdgcn_s_wait_asynccnt(n)
# else
#  define WAIT_ASYNC(n) asm volatile("s_wait_asynccnt %0" :: "n"(n) : "memory")
# endif
#else
# define WAIT_ASYNC(n) do {} while (0)
#endif

#if defined(__AMDGCN__) && __has_builtin(__builtin_amdgcn_exp2f)
#define EXP2F(x) __builtin_amdgcn_exp2f(x)
#else
#define EXP2F(x) exp2f(x)
#endif

// A fragment, 16-bit 16x32, K contiguous (matches B):
//   lanes 0-15  : K = kc+0..15
//   lanes 16-31 : K = kc+16..31
__device__ __forceinline__ v16h load_a_frag(const _Float16* p0, int lane) {
  return *(const v16h*)(p0 + ((lane >> 4) << 4));
}

// B fragment, 16-bit 32x16: lane = one column N; K contiguous (0..15 / 16..31).
__device__ __forceinline__ v16h load_b_frag(const _Float16* p0, int lane) {
  return *(const v16h*)(p0 + ((lane >> 4) << 4));
}

#define WMMA_F16(A, B, C) \
  __builtin_amdgcn_wmma_f32_16x16x32_f16(false, (A), false, (B), (short)0, (C), false, false)

// ---------------------------------------------------------------------------
// Prep kernel: one-time f32 -> f16 conversion of K (transposed to [bh][s][c],
// pre-scaled) and V (kept [bh][c][s]). These layouts match the LDS fragment
// layouts byte-for-byte, so the FA kernel can async-DMA tiles straight to LDS.
// ---------------------------------------------------------------------------
__global__ __launch_bounds__(256) void qkv_prep_kernel(
    const float* __restrict__ qkv, _Float16* __restrict__ k16,
    _Float16* __restrict__ v16) {
  __shared__ float tile[CH * 65];
  const int tid = threadIdx.x;
  const int bh = blockIdx.x;
  const int b  = bh >> 4;
  const int h  = bh & 15;
  const int s0 = blockIdx.y * 64;

  const float* kbase = qkv + ((size_t)b * 3 * NHEADS * CH + (size_t)NHEADS * CH
                              + (size_t)h * CH) * LEN;
  const float* vbase = kbase + (size_t)NHEADS * CH * LEN;

  _Float16* vdst = v16 + (size_t)bh * CH * LEN;
  for (int idx = tid; idx < CH * 64; idx += 256) {
    int c = idx >> 6;
    int s = idx & 63;
    tile[c * 65 + s] = kbase[c * LEN + s0 + s];              // stage K tile
    vdst[(size_t)c * LEN + s0 + s] = (_Float16)vbase[c * LEN + s0 + s];
  }
  __syncthreads();

  _Float16* kdst = k16 + (size_t)bh * LEN * CH;
  for (int idx = tid; idx < CH * 64; idx += 256) {
    int s = idx >> 6;
    int c = idx & 63;                                        // c fastest: coalesced
    kdst[(size_t)(s0 + s) * CH + c] = (_Float16)(tile[c * 65 + s] * QSCALE);
  }
}

// Issue one K/V tile copy (8 KB each) into LDS. With async support this is
// 4 global_load_async_to_lds_b128 per thread tracked on ASYNCcnt.
__device__ __forceinline__ void issue_tile_copy(
    const _Float16* __restrict__ kg, const _Float16* __restrict__ vg,
    _Float16* kt, _Float16* vt, int s0, int tid) {
  const int r0 = tid >> 3;          // 8 lanes x 16B cover one 128B row
  const int o  = (tid & 7) * 8;     // half offset within row
#pragma unroll
  for (int rr = r0; rr < BN; rr += 32) {
    const _Float16* gk = kg + (size_t)(s0 + rr) * CH + o;
    const _Float16* gv = vg + (size_t)rr * LEN + s0 + o;
    _Float16* lk = kt + rr * LDSS + o;
    _Float16* lv = vt + rr * LDSS + o;
#if HAVE_ASYNC
    __builtin_amdgcn_global_load_async_to_lds_b128(
        (__attribute__((address_space(1))) v4i*)(void*)(const void*)gk,
        (__attribute__((address_space(3))) v4i*)(void*)lk, 0, 0);
    __builtin_amdgcn_global_load_async_to_lds_b128(
        (__attribute__((address_space(1))) v4i*)(void*)(const void*)gv,
        (__attribute__((address_space(3))) v4i*)(void*)lv, 0, 0);
#else
    *(v8h*)lk = *(const v8h*)gk;
    *(v8h*)lv = *(const v8h*)gv;
#endif
  }
}

// ---------------------------------------------------------------------------
// Flash-attention kernel. PRE=true: stream pre-converted f16 K/V tiles with
// double-buffered async LDS copies. PRE=false: inline f32->f16 conversion.
// ---------------------------------------------------------------------------
template <bool PRE>
__global__ __launch_bounds__(256) void qkv_attn_fa_kernel(
    const float* __restrict__ qkv, const _Float16* __restrict__ k16,
    const _Float16* __restrict__ v16, float* __restrict__ out) {
  __shared__ _Float16 Qh[BM * LDSS];          // Q tile [t][c], pre-scaled
  __shared__ _Float16 KT[2][BN * LDSS];       // K tiles [s][c], double buffered
  __shared__ _Float16 VT[2][CH * LDSS];       // V tiles [c][s], double buffered
  __shared__ _Float16 Ps[8 * 16 * LDSS];      // per-wave P tile [t][s]

  const int tid  = threadIdx.x;
  const int lane = tid & 31;
  const int wave = tid >> 5;
  const int lrow = lane & 15;
  const int half8 = (lane >> 4) << 3;

  const int bh = blockIdx.x;
  const int b  = bh >> 4;
  const int h  = bh & 15;
  const int t0 = blockIdx.y * BM;

  const float* qbase = qkv + ((size_t)b * 3 * NHEADS * CH + (size_t)h * CH) * LEN;
  const float* kbase = qbase + (size_t)NHEADS * CH * LEN;
  const float* vbase = qbase + (size_t)2 * NHEADS * CH * LEN;
  const _Float16* kg = PRE ? k16 + (size_t)bh * LEN * CH : nullptr;
  const _Float16* vg = PRE ? v16 + (size_t)bh * CH * LEN : nullptr;

  // ---- Stage Q tile: global [c][t] f32 -> LDS [t][c] f16 (pre-scaled) ----
  for (int idx = tid; idx < CH * BM; idx += 256) {
    int c = idx >> 7;
    int t = idx & (BM - 1);
    Qh[t * LDSS + c] = (_Float16)(qbase[c * LEN + t0 + t] * QSCALE);
  }

  if constexpr (PRE) issue_tile_copy(kg, vg, KT[0], VT[0], 0, tid);

  __syncthreads();
  const int m0 = wave * 16;
  const v16h qa0 = load_a_frag(&Qh[(m0 + lrow) * LDSS + 0],  lane);
  const v16h qa1 = load_a_frag(&Qh[(m0 + lrow) * LDSS + 32], lane);

  v8f o0 = {}, o1 = {}, o2 = {}, o3 = {};
  float m_run[8], l_run[8];
#pragma unroll
  for (int i = 0; i < 8; ++i) { m_run[i] = -INFINITY; l_run[i] = 0.f; }

  _Float16* pbuf = &Ps[wave * 16 * LDSS];

  // ================= S-tile loop =================
  for (int it = 0; it < NTILES; ++it) {
    const int s0 = it * BN;
    __syncthreads();   // all waves done with compute of tile it-1
    if constexpr (PRE) {
      if (it + 1 < NTILES) {
        issue_tile_copy(kg, vg, KT[(it + 1) & 1], VT[(it + 1) & 1],
                        s0 + BN, tid);
        WAIT_ASYNC(4);                 // tile `it` (4 older async ops) done
      } else {
        WAIT_ASYNC(0);
      }
    } else {
      for (int idx = tid; idx < CH * BN; idx += 256) {
        int c = idx >> 6;
        int s = idx & (BN - 1);
        KT[0][s * LDSS + c] = (_Float16)(kbase[c * LEN + s0 + s] * QSCALE);
        VT[0][c * LDSS + s] = (_Float16)vbase[c * LEN + s0 + s];
      }
    }
    __syncthreads();   // tile `it` visible to all waves

    const _Float16* Ktc = PRE ? KT[it & 1] : KT[0];
    const _Float16* Vtc = PRE ? VT[it & 1] : VT[0];

    // ---- W = Q^T K (16 x 64 per wave), f32 accum, log2 domain ----
    v8f w0 = {}, w1 = {}, w2 = {}, w3 = {};
#define QK_TILE(WACC, NC)                                                  \
    do {                                                                   \
      v16h bk0 = load_b_frag(&Ktc[((NC)*16 + lrow) * LDSS + 0],  lane);    \
      v16h bk1 = load_b_frag(&Ktc[((NC)*16 + lrow) * LDSS + 32], lane);    \
      WACC = WMMA_F16(qa0, bk0, WACC);                                     \
      WACC = WMMA_F16(qa1, bk1, WACC);                                     \
    } while (0)
    QK_TILE(w0, 0); QK_TILE(w1, 1); QK_TILE(w2, 2); QK_TILE(w3, 3);
#undef QK_TILE

    // ---- Online softmax: row = (vgpr i, lane half), N across 16 lanes ----
    float mt[8];
#pragma unroll
    for (int i = 0; i < 8; ++i)
      mt[i] = fmaxf(fmaxf(w0[i], w1[i]), fmaxf(w2[i], w3[i]));
#pragma unroll
    for (int off = 8; off >= 1; off >>= 1)
#pragma unroll
      for (int i = 0; i < 8; ++i)
        mt[i] = fmaxf(mt[i], __shfl_xor(mt[i], off, 32));

    float rs[8];
#pragma unroll
    for (int i = 0; i < 8; ++i) {
      float mnew = fmaxf(m_run[i], mt[i]);
      float fac  = EXP2F(m_run[i] - mnew);
      m_run[i] = mnew;
      w0[i] = EXP2F(w0[i] - mnew);
      w1[i] = EXP2F(w1[i] - mnew);
      w2[i] = EXP2F(w2[i] - mnew);
      w3[i] = EXP2F(w3[i] - mnew);
      rs[i] = (w0[i] + w1[i]) + (w2[i] + w3[i]);
      o0[i] *= fac; o1[i] *= fac; o2[i] *= fac; o3[i] *= fac;
      l_run[i] *= fac;
    }
#pragma unroll
    for (int off = 8; off >= 1; off >>= 1)
#pragma unroll
      for (int i = 0; i < 8; ++i)
        rs[i] += __shfl_xor(rs[i], off, 32);
#pragma unroll
    for (int i = 0; i < 8; ++i) l_run[i] += rs[i];

    // ---- P -> f16 per-wave LDS [t][s] (same-wave LDS ops are in-order) ----
#pragma unroll
    for (int i = 0; i < 8; ++i) {
      _Float16* pp = &pbuf[(i + half8) * LDSS + lrow];
      pp[0]  = (_Float16)w0[i];
      pp[16] = (_Float16)w1[i];
      pp[32] = (_Float16)w2[i];
      pp[48] = (_Float16)w3[i];
    }

    // ---- O += P * V^T (K dim = s, N dim = c) ----
    v16h pa0 = load_a_frag(&pbuf[lrow * LDSS + 0],  lane);
    v16h pa1 = load_a_frag(&pbuf[lrow * LDSS + 32], lane);
#define PV_TILE(OACC, NC)                                                  \
    do {                                                                   \
      v16h bv0 = load_b_frag(&Vtc[((NC)*16 + lrow) * LDSS + 0],  lane);    \
      v16h bv1 = load_b_frag(&Vtc[((NC)*16 + lrow) * LDSS + 32], lane);    \
      OACC = WMMA_F16(pa0, bv0, OACC);                                     \
      OACC = WMMA_F16(pa1, bv1, OACC);                                     \
    } while (0)
    PV_TILE(o0, 0); PV_TILE(o1, 1); PV_TILE(o2, 2); PV_TILE(o3, 3);
#undef PV_TILE
  }

  // ---- Epilogue: O[t][c]/l -> out[b][h*64+c][t]; 8 consecutive t per lane ----
  float linv[8];
#pragma unroll
  for (int i = 0; i < 8; ++i) linv[i] = 1.0f / l_run[i];

  float* obase = out + ((size_t)b * NHEADS * CH + (size_t)h * CH) * LEN;
  const int tg = t0 + m0 + half8;

#define STORE_O(OACC, NC)                                                  \
  do {                                                                     \
    int c = (NC) * 16 + lrow;                                              \
    v4f lo = { OACC[0]*linv[0], OACC[1]*linv[1],                           \
               OACC[2]*linv[2], OACC[3]*linv[3] };                         \
    v4f hi = { OACC[4]*linv[4], OACC[5]*linv[5],                           \
               OACC[6]*linv[6], OACC[7]*linv[7] };                         \
    *(v4f*)(&obase[(size_t)c * LEN + tg])     = lo;                        \
    *(v4f*)(&obase[(size_t)c * LEN + tg + 4]) = hi;                        \
  } while (0)
  STORE_O(o0, 0); STORE_O(o1, 1); STORE_O(o2, 2); STORE_O(o3, 3);
#undef STORE_O
}

extern "C" void kernel_launch(void* const* d_in, const int* in_sizes, int n_in,
                              void* d_out, int out_size, void* d_ws, size_t ws_size,
                              hipStream_t stream) {
  (void)in_sizes; (void)n_in; (void)out_size;
  const float* qkv = (const float*)d_in[0];
  float* out = (float*)d_out;
  dim3 block(256);

  const size_t need = (size_t)2 * 32 * LEN * CH * sizeof(_Float16); // K16 + V16
  if (ws_size >= need) {
    _Float16* k16 = (_Float16*)d_ws;
    _Float16* v16 = k16 + (size_t)32 * LEN * CH;
    qkv_prep_kernel<<<dim3(32, LEN / 64), block, 0, stream>>>(qkv, k16, v16);
    qkv_attn_fa_kernel<true><<<dim3(32, LEN / BM), block, 0, stream>>>(
        qkv, k16, v16, out);
  } else {
    qkv_attn_fa_kernel<false><<<dim3(32, LEN / BM), block, 0, stream>>>(
        qkv, nullptr, nullptr, out);
  }
}
